// Model1_new_23983097380900
// MI455X (gfx1250) — matrix operand order repaired and checked by reading, not compiled, not run on hardware
//
#include <hip/hip_runtime.h>
#include <hip/hip_bf16.h>
#include <math.h>

// ---------------------------------------------------------------------------
// Types for CDNA5 WMMA (gfx1250, wave32)
// ---------------------------------------------------------------------------
typedef __attribute__((ext_vector_type(16))) __bf16 v16bf;
typedef __attribute__((ext_vector_type(8)))  float  v8f;

__device__ __forceinline__ float gelu_erf(float x) {
    return 0.5f * x * (1.0f + erff(x * 0.70710678118654752440f));
}

// 16 fp32 -> v16bf; lowers to v_cvt_pk_bf16_f32 on gfx1250.
__device__ __forceinline__ v16bf cvt16(float4 a, float4 b, float4 c, float4 d) {
    v16bf r;
    r[0]  = (__bf16)a.x; r[1]  = (__bf16)a.y; r[2]  = (__bf16)a.z; r[3]  = (__bf16)a.w;
    r[4]  = (__bf16)b.x; r[5]  = (__bf16)b.y; r[6]  = (__bf16)b.z; r[7]  = (__bf16)b.w;
    r[8]  = (__bf16)c.x; r[9]  = (__bf16)c.y; r[10] = (__bf16)c.z; r[11] = (__bf16)c.w;
    r[12] = (__bf16)d.x; r[13] = (__bf16)d.y; r[14] = (__bf16)d.z; r[15] = (__bf16)d.w;
    return r;
}

// A-fragment (16x32 bf16): lane holds row m = lane&15,
// 16 consecutive K values starting at (lane>>4)*16 (= 2*abase), matching B.
__device__ __forceinline__ v16bf load_a_frag(const float* __restrict__ p, int abase) {
    const float4* q0 = (const float4*)(p + 2 * abase);
    const float4* q1 = (const float4*)(p + 2 * abase + 8);
    return cvt16(q0[0], q0[1], q1[0], q1[1]);
}

// B-fragment (32x16 bf16): lane holds column n = lane&15, 16 consecutive K
// values starting at (lane>>4)*16.
__device__ __forceinline__ v16bf load_b_frag(const float* __restrict__ p) {
    const float4* q = (const float4*)p;
    return cvt16(q[0], q[1], q[2], q[3]);
}

// ---------------------------------------------------------------------------
// Generic GEMM with split-K: for K-partition z = blockIdx.z:
//   Cz = epilogue( scale * (A[:, zK:(z+1)K] @ B[:, zK:(z+1)K]^T) + bias )
//   stored at C + z*cPartStride.
// A: [M,Ktot] f32 row-major (lda), B: [N,Ktot] f32 row-major (ldb)
// Block: 256 thr = 8 waves (2 in M x 4 in N), wave tile 32x64,
// block tile 64(M) x 256(N).  Requires M%64==0, N%256==0, K%32==0.
// ---------------------------------------------------------------------------
__global__ void __launch_bounds__(256)
gemm_bf16_wmma(const float* __restrict__ A, const float* __restrict__ B,
               const float* __restrict__ bias, float* __restrict__ C,
               int K, int lda, int ldb, int ldc,
               float scale, int applyGelu, int transStore, long cPartStride) {
    const int lane  = threadIdx.x & 31;
    const int wid   = threadIdx.x >> 5;
    const int waveM = wid >> 2;          // 0..1
    const int waveN = wid & 3;           // 0..3
    const int mBase = blockIdx.y * 64 + waveM * 32;
    const int nBase = blockIdx.x * 256 + waveN * 64;
    const int lrow  = lane & 15;
    const int khalf = lane >> 4;
    const int abase = khalf * 8;

    const long k0   = (long)blockIdx.z * K;   // split-K partition start
    const long kEnd = k0 + K;
    C += (long)blockIdx.z * cPartStride;

    const float* Ar0 = A + (long)(mBase + lrow)      * lda;
    const float* Ar1 = A + (long)(mBase + 16 + lrow) * lda;
    const float* Br0 = B + (long)(nBase + lrow)      * ldb + khalf * 16;
    const float* Br1 = B + (long)(nBase + 16 + lrow) * ldb + khalf * 16;
    const float* Br2 = B + (long)(nBase + 32 + lrow) * ldb + khalf * 16;
    const float* Br3 = B + (long)(nBase + 48 + lrow) * ldb + khalf * 16;

    v8f acc[2][4];
    #pragma unroll
    for (int i = 0; i < 2; ++i)
        #pragma unroll
        for (int j = 0; j < 4; ++j)
            acc[i][j] = (v8f){0.f, 0.f, 0.f, 0.f, 0.f, 0.f, 0.f, 0.f};

    for (long k = k0; k < kEnd; k += 32) {
        v16bf a0 = load_a_frag(Ar0 + k, abase);
        v16bf a1 = load_a_frag(Ar1 + k, abase);
        v16bf b0 = load_b_frag(Br0 + k);
        v16bf b1 = load_b_frag(Br1 + k);
        v16bf b2 = load_b_frag(Br2 + k);
        v16bf b3 = load_b_frag(Br3 + k);
        acc[0][0] = __builtin_amdgcn_wmma_f32_16x16x32_bf16(false, a0, false, b0,
                                                            (short)0, acc[0][0], false, false);
        acc[0][1] = __builtin_amdgcn_wmma_f32_16x16x32_bf16(false, a0, false, b1,
                                                            (short)0, acc[0][1], false, false);
        acc[0][2] = __builtin_amdgcn_wmma_f32_16x16x32_bf16(false, a0, false, b2,
                                                            (short)0, acc[0][2], false, false);
        acc[0][3] = __builtin_amdgcn_wmma_f32_16x16x32_bf16(false, a0, false, b3,
                                                            (short)0, acc[0][3], false, false);
        acc[1][0] = __builtin_amdgcn_wmma_f32_16x16x32_bf16(false, a1, false, b0,
                                                            (short)0, acc[1][0], false, false);
        acc[1][1] = __builtin_amdgcn_wmma_f32_16x16x32_bf16(false, a1, false, b1,
                                                            (short)0, acc[1][1], false, false);
        acc[1][2] = __builtin_amdgcn_wmma_f32_16x16x32_bf16(false, a1, false, b2,
                                                            (short)0, acc[1][2], false, false);
        acc[1][3] = __builtin_amdgcn_wmma_f32_16x16x32_bf16(false, a1, false, b3,
                                                            (short)0, acc[1][3], false, false);
    }

    #pragma unroll
    for (int mt = 0; mt < 2; ++mt) {
        #pragma unroll
        for (int nt = 0; nt < 4; ++nt) {
            const int gn = nBase + nt * 16 + lrow;
            const float bv = bias ? bias[gn] : 0.0f;
            v8f a = acc[mt][nt];
            if (!transStore) {
                #pragma unroll
                for (int v = 0; v < 8; ++v) {
                    const int gm = mBase + mt * 16 + 8 * khalf + v;  // C layout: M = v + 8*khalf
                    float val = a[v] * scale + bv;
                    if (applyGelu) val = gelu_erf(val);
                    C[(long)gm * ldc + gn] = val;
                }
            } else {
                float o[8];
                #pragma unroll
                for (int v = 0; v < 8; ++v) {
                    float val = a[v] * scale + bv;
                    if (applyGelu) val = gelu_erf(val);
                    o[v] = val;
                }
                float4* q = (float4*)(C + (long)gn * ldc + mBase + mt * 16 + 8 * khalf);
                q[0] = make_float4(o[0], o[1], o[2], o[3]);
                q[1] = make_float4(o[4], o[5], o[6], o[7]);
            }
        }
    }
}

// ---------------------------------------------------------------------------
// Split-K reduction: C[idx] = epi( sum_p parts[p*stride + idx] + bias[j] )
// Output is [M,N] row-major with ldc == N, N a power of two (nMask = N-1).
// Fixed summation order => bitwise deterministic.
// ---------------------------------------------------------------------------
__global__ void __launch_bounds__(256)
reduce_parts_kernel(const float* __restrict__ parts, int nParts, long stride,
                    const float* __restrict__ bias, float* __restrict__ C,
                    int nMask, int applyGelu) {
    const long idx = (long)blockIdx.x * 256 + threadIdx.x;
    float s = 0.0f;
    for (int p = 0; p < nParts; ++p)
        s += parts[(long)p * stride + idx];
    if (bias) s += bias[idx & nMask];
    if (applyGelu) s = gelu_erf(s);
    C[idx] = s;
}

// ---------------------------------------------------------------------------
// Row-center: Xc[i,:] = X[i,:] - mean(X[i,:]).  One block per row (n=256,d=4096)
// ---------------------------------------------------------------------------
__global__ void __launch_bounds__(256)
rowcenter_kernel(const float* __restrict__ X, float* __restrict__ Xc) {
    __shared__ float red[256];
    const int row = blockIdx.x, t = threadIdx.x;
    const float4* xr = (const float4*)(X + (long)row * 4096);
    float4 v0 = xr[t], v1 = xr[t + 256], v2 = xr[t + 512], v3 = xr[t + 768];
    float s = v0.x + v0.y + v0.z + v0.w + v1.x + v1.y + v1.z + v1.w
            + v2.x + v2.y + v2.z + v2.w + v3.x + v3.y + v3.z + v3.w;
    red[t] = s;
    __syncthreads();
    for (int o = 128; o > 0; o >>= 1) {
        if (t < o) red[t] += red[t + o];
        __syncthreads();
    }
    const float mean = red[0] * (1.0f / 4096.0f);
    float4* out = (float4*)(Xc + (long)row * 4096);
    v0.x -= mean; v0.y -= mean; v0.z -= mean; v0.w -= mean;
    v1.x -= mean; v1.y -= mean; v1.z -= mean; v1.w -= mean;
    v2.x -= mean; v2.y -= mean; v2.z -= mean; v2.w -= mean;
    v3.x -= mean; v3.y -= mean; v3.z -= mean; v3.w -= mean;
    out[t] = v0; out[t + 256] = v1; out[t + 512] = v2; out[t + 768] = v3;
}

// ---------------------------------------------------------------------------
// weight_net: w[l] = Linear(GELU(Linear(lag)))  for lag = -64..64 (129 values)
// ---------------------------------------------------------------------------
__global__ void weight_net_kernel(const float* __restrict__ w1, const float* __restrict__ b1,
                                  const float* __restrict__ w2, const float* __restrict__ b2,
                                  float* __restrict__ w) {
    const int l = threadIdx.x;
    if (l >= 129) return;
    const float lag = (float)(l - 64);
    float acc = b2[0];
    #pragma unroll 8
    for (int j = 0; j < 64; ++j)
        acc += gelu_erf(lag * w1[j] + b1[j]) * w2[j];
    w[l] = acc;
}

// ---------------------------------------------------------------------------
// Banded-Toeplitz apply (the collapsed 65-GEMM covariance):
//   XcB[i,s] = sum_{l=-64}^{64} w[64+l] * Xc[i, s-l]   (skip out-of-range)
// ---------------------------------------------------------------------------
__global__ void __launch_bounds__(256)
conv_lags_kernel(const float* __restrict__ Xc, const float* __restrict__ w,
                 float* __restrict__ out) {
    __shared__ float ws[129];
    const int t = threadIdx.x;
    if (t < 129) ws[t] = w[t];
    __syncthreads();
    const int idx = blockIdx.x * 256 + t;         // 256*4096 outputs
    const int i = idx >> 12;
    const int s = idx & 4095;
    const float* row = Xc + (long)i * 4096;
    int lmin = s - 4095; if (lmin < -64) lmin = -64;
    int lmax = s;        if (lmax >  64) lmax =  64;
    float acc = 0.0f;
    for (int l = lmin; l <= lmax; ++l)
        acc += ws[l + 64] * row[s - l];
    out[idx] = acc;
}

// ---------------------------------------------------------------------------
// Transpose [256,4096] -> [4096,256]
// ---------------------------------------------------------------------------
__global__ void transpose_kernel(const float* __restrict__ X, float* __restrict__ Xt) {
    __shared__ float tile[32][33];
    const int tx = threadIdx.x;               // 0..31
    const int x = blockIdx.x * 32 + tx;       // col of X
    const int y0 = blockIdx.y * 32;           // row base of X
    for (int j = threadIdx.y; j < 32; j += 8)
        tile[j][tx] = X[(long)(y0 + j) * 4096 + x];
    __syncthreads();
    for (int j = threadIdx.y; j < 32; j += 8)
        Xt[(long)(blockIdx.x * 32 + j) * 256 + y0 + tx] = tile[tx][j];
}

// Copy X into left half of z (z is [256, 8192])
__global__ void __launch_bounds__(256)
copy_to_z_kernel(const float* __restrict__ X, float* __restrict__ z) {
    const int idx = blockIdx.x * 256 + threadIdx.x;   // over 262144 float4's
    const int i = idx >> 10;                          // 1024 float4 per row
    const int j = idx & 1023;
    ((float4*)(z + (long)i * 8192))[j] = ((const float4*)(X + (long)i * 4096))[j];
}

// ---------------------------------------------------------------------------
// Launch
// ---------------------------------------------------------------------------
extern "C" void kernel_launch(void* const* d_in, const int* in_sizes, int n_in,
                              void* d_out, int out_size, void* d_ws, size_t ws_size,
                              hipStream_t stream) {
    const float* X     = (const float*)d_in[0];   // [256,4096]
    const float* wn_w1 = (const float*)d_in[1];   // [1,64]
    const float* wn_b1 = (const float*)d_in[2];   // [64]
    const float* wn_w2 = (const float*)d_in[3];   // [64,1]
    const float* wn_b2 = (const float*)d_in[4];   // [1]
    const float* fc1_w = (const float*)d_in[5];   // [4096,8192]
    const float* fc1_b = (const float*)d_in[6];   // [4096]
    const float* fc2_w = (const float*)d_in[7];   // [4096,4096]
    const float* fc2_b = (const float*)d_in[8];   // [4096]
    const float* fc3_w = (const float*)d_in[9];   // [4096,4096]
    const float* fc3_b = (const float*)d_in[10];  // [4096]
    const float* proj  = (const float*)d_in[11];  // [512,256]
    float* out = (float*)d_out;                   // [512,4096]

    const int n = 256, d = 4096, hid = 4096, cin = 8192, y0 = 512, y1 = 4096;

    float* ws    = (float*)d_ws;
    float* Xc    = ws;                             // 256*4096
    float* Xt    = Xc  + (size_t)n * d;            // 4096*256
    float* XcB   = Xt  + (size_t)d * n;            // 256*4096 (later reused as h3)
    float* wlag  = XcB + (size_t)n * d;            // 256 (129 used)
    float* cov   = wlag + 256;                     // 256*256
    float* z     = cov + (size_t)n * n;            // 256*8192
    float* h1    = z  + (size_t)n * cin;           // 256*4096
    float* h2    = h1 + (size_t)n * hid;           // 256*4096
    float* o3T   = h2 + (size_t)n * hid;           // 4096*256 (transposed fc3 out)
    float* parts = o3T + (size_t)d * n;            // 2*256*4096 split-K partials

    // 1) Xc = X - rowmean(X)
    rowcenter_kernel<<<n, 256, 0, stream>>>(X, Xc);
    // 2) lag weights via tiny MLP (erf-GELU)
    weight_net_kernel<<<1, 160, 0, stream>>>(wn_w1, wn_b1, wn_w2, wn_b2, wlag);
    // 3) banded Toeplitz apply (replaces 65 gamma GEMMs)
    conv_lags_kernel<<<(n * d) / 256, 256, 0, stream>>>(Xc, wlag, XcB);
    // 4) Xt = X^T (so cov@X fits the A@B^T kernel)
    transpose_kernel<<<dim3(d / 32, n / 32), dim3(32, 8), 0, stream>>>(X, Xt);
    // 5) z[:, :d] = X
    copy_to_z_kernel<<<(n * d / 4) / 256, 256, 0, stream>>>(X, z);

    // 6) cov = (1/d) * XcB @ Xc^T   [256,256]  -- 16-way split-K (4 -> 64 blocks)
    gemm_bf16_wmma<<<dim3(n / 256, n / 64, 16), 256, 0, stream>>>(
        XcB, Xc, nullptr, parts, d / 16, d, d, n, 1.0f / (float)d, 0, 0, (long)n * n);
    reduce_parts_kernel<<<(n * n) / 256, 256, 0, stream>>>(
        parts, 16, (long)n * n, nullptr, cov, n - 1, 0);

    // 7) z[:, d:] = cov @ X = cov @ Xt^T   [256,4096], ldc=8192, col offset d
    gemm_bf16_wmma<<<dim3(d / 256, n / 64, 1), 256, 0, stream>>>(
        cov, Xt, nullptr, z + d, n, n, n, cin, 1.0f, 0, 0, 0);

    // 8) h1 = gelu(z @ fc1_w^T + fc1_b)   [256,4096], K=8192 -- 2-way split-K
    gemm_bf16_wmma<<<dim3(hid / 256, n / 64, 2), 256, 0, stream>>>(
        z, fc1_w, nullptr, parts, cin / 2, cin, cin, hid, 1.0f, 0, 0, (long)n * hid);
    reduce_parts_kernel<<<(n * hid) / 256, 256, 0, stream>>>(
        parts, 2, (long)n * hid, fc1_b, h1, hid - 1, 1);

    // 9) h2 = gelu(h1 @ fc2_w^T + fc2_b)  [256,4096], K=4096 -- 2-way split-K
    gemm_bf16_wmma<<<dim3(hid / 256, n / 64, 2), 256, 0, stream>>>(
        h1, fc2_w, nullptr, parts, hid / 2, hid, hid, hid, 1.0f, 0, 0, (long)n * hid);
    reduce_parts_kernel<<<(n * hid) / 256, 256, 0, stream>>>(
        parts, 2, (long)n * hid, fc2_b, h2, hid - 1, 1);

    // 10) h3 = h2 @ fc3_w^T + fc3_b  [256,4096] -- 2-way split-K, reuse XcB as h3
    float* h3 = XcB;
    gemm_bf16_wmma<<<dim3(y1 / 256, n / 64, 2), 256, 0, stream>>>(
        h2, fc3_w, nullptr, parts, hid / 2, hid, hid, y1, 1.0f, 0, 0, (long)n * y1);
    reduce_parts_kernel<<<(n * y1) / 256, 256, 0, stream>>>(
        parts, 2, (long)n * y1, fc3_b, h3, y1 - 1, 0);
    //     o3T = h3^T  [4096,256]
    transpose_kernel<<<dim3(y1 / 32, n / 32), dim3(32, 8), 0, stream>>>(h3, o3T);

    // 11) out = proj @ o3T^T   [512,4096], K=256
    gemm_bf16_wmma<<<dim3(y1 / 256, y0 / 64, 1), 256, 0, stream>>>(
        proj, o3T, nullptr, out, n, n, n, y1, 1.0f, 0, 0, 0);

    (void)in_sizes; (void)n_in; (void)out_size; (void)ws_size;
}
